// MMoE_86698209837570
// MI455X (gfx1250) — matrix operand order repaired and checked by reading, not compiled, not run on hardware
//
#include <hip/hip_runtime.h>
#include <cstdint>

// ---------------------------------------------------------------------------
// MMoE / DeepFM forward, MI455X (gfx1250, wave32, WMMA 16x16x32 f16 -> f32 acc)
// Register-blocked (one A-fragment feeds up to 9 WMMA accumulators); all
// wave-level branching is scalarized (readfirstlane) and hoisted out of the
// K-loops so the hot loops are straight-line WMMA + clause'd b128 loads.
// ---------------------------------------------------------------------------

typedef _Float16 half_t;
typedef __attribute__((ext_vector_type(16))) _Float16 v16h;
typedef __attribute__((ext_vector_type(8)))  float    v8f;

#define BATCH   16384
#define DREAL   464     // 8*50 emb + 64 cont
#define DPAD    480     // K padded to multiple of 32
#define COPAD   512     // co-GEMM N padded so 8 waves x 4 tiles cover it evenly
#define H1      256
#define H2      128
#define NEXP    4
#define NTASK   3

union Frag16 { v16h v; uint4 q[2]; };

__device__ __forceinline__ v8f wmma16(v16h a, v16h b, v8f c) {
  // v_wmma_f32_16x16x32_f16: D = A(16x32 f16) * B(32x16 f16) + C(16x16 f32)
  return __builtin_amdgcn_wmma_f32_16x16x32_f16(false, a, false, b, (short)0, c,
                                                false, false);
}

// A fragment (row-major A[16][stride]): lane m = lane&15, grp = lane>>4.
// halves 0..7  hold K = kbase + grp*8 + j ; halves 8..15 hold +16.
__device__ __forceinline__ v16h load_a_frag(const half_t* base, int stride,
                                            int m, int grp, int kbase) {
  const half_t* p = base + (size_t)m * stride + kbase + grp * 8;
  Frag16 f;
  f.q[0] = *(const uint4*)(p);
  f.q[1] = *(const uint4*)(p + 16);
  return f.v;
}

// B fragment from a precomputed per-lane row pointer (weights transposed
// W_t[N][K]): halves 0..7 hold K = kbase + grp*8 + j ; halves 8..15 hold +16.
__device__ __forceinline__ v16h load_b_row(const half_t* rowp, int grp,
                                           int kbase) {
  const half_t* p = rowp + kbase + grp * 8;
  Frag16 f;
  f.q[0] = *(const uint4*)(p);
  f.q[1] = *(const uint4*)(p + 16);
  return f.v;
}

// ---------------------------------------------------------------------------
// Weight convert: fp32 [batch][Ksrc][Nsrc] -> f16 transposed [batch*Nd][Kpad],
// zero-filling k >= Ksrc and n >= Nsrc (makes all padding inert).
// ---------------------------------------------------------------------------
__global__ void convert_weights(const float* __restrict__ src,
                                half_t* __restrict__ dst,
                                int Nd, int Ns, int Kpad, int Ks, int total) {
  int d = blockIdx.x * blockDim.x + threadIdx.x;
  if (d >= total) return;
  int k  = d % Kpad;
  int rn = d / Kpad;
  int n  = rn % Nd;
  int b  = rn / Nd;
  float v = (k < Ks && n < Ns) ? src[((size_t)b * Ks + k) * Ns + n] : 0.0f;
  dst[d] = (half_t)v;
}

// ---------------------------------------------------------------------------
// Stage 1: gather embeddings + continuous relu layer -> xh f16 [B][480]
// ---------------------------------------------------------------------------
__global__ __launch_bounds__(128) void build_x(
    const int* __restrict__ disc, const float* __restrict__ cont,
    const float* __restrict__ t0, const float* __restrict__ t1,
    const float* __restrict__ t2, const float* __restrict__ t3,
    const float* __restrict__ t4, const float* __restrict__ t5,
    const float* __restrict__ t6, const float* __restrict__ t7,
    const float* __restrict__ contW, const float* __restrict__ contB,
    half_t* __restrict__ xh) {
  int b = blockIdx.x, t = threadIdx.x;
  const int* drow = disc + b * 8;
  half_t* xrow = xh + (size_t)b * DPAD;
  for (int c = t; c < 400; c += 128) {
    int i = c / 50, j = c - i * 50;
    const float* tab;
    switch (i) {
      case 0: tab = t0; break; case 1: tab = t1; break;
      case 2: tab = t2; break; case 3: tab = t3; break;
      case 4: tab = t4; break; case 5: tab = t5; break;
      case 6: tab = t6; break; default: tab = t7; break;
    }
    xrow[c] = (half_t)tab[(size_t)drow[i] * 50 + j];
  }
  if (t < 64) {                       // cont layer: tiny VALU work (134 MFLOP)
    float acc = contB[t];
    const float* crow = cont + (size_t)b * 64;
    #pragma unroll 8
    for (int k = 0; k < 64; ++k) acc = fmaf(crow[k], contW[k * 64 + t], acc);
    xrow[400 + t] = (half_t)fmaxf(acc, 0.0f);
  } else if (t < 80) {
    xrow[400 + t] = (half_t)0.0f;     // zero K-pad cols 464..479
  }
}

// ---------------------------------------------------------------------------
// Stage 2: cross network + FM -> h -> hh f16 [B][480].
// Block = 16 rows, 8 waves; each wave register-blocks N-tiles on one A load.
// ---------------------------------------------------------------------------
__global__ __launch_bounds__(256) void cross_fm(
    const half_t* __restrict__ xh,
    const float* __restrict__ cb1, const float* __restrict__ cb2,
    const float* __restrict__ cob,
    const half_t* __restrict__ cW1t, const half_t* __restrict__ cW2t,
    const half_t* __restrict__ coWt, const half_t* __restrict__ fmWt,
    half_t* __restrict__ hh) {
  __shared__ __align__(16) half_t lds_c1[16][H1];   // relu(x@cW1)
  __shared__ __align__(16) half_t lds_c2[16][H2];   // relu(c1@cW2)
  __shared__ float lds_xe[16][16];                  // x@fm_W (cols>=10 are 0)
  __shared__ float lds_fm[16];

  const int b0   = blockIdx.x * 16;
  // Scalar (SGPR) wave id: all wave-level control flow is s_cbranch, no EXEC
  // manipulation around WMMAs.
  const int wv   = __builtin_amdgcn_readfirstlane(threadIdx.x) >> 5;
  const int lane = threadIdx.x & 31;
  const int grp  = lane >> 4;
  const int mn   = lane & 15;
  const half_t* arow = xh + (size_t)b0 * DPAD;

  // GEMM1: c1[16,256] = relu(x @ cW1 + cb1), K=480.  Wave w owns N-tiles
  // {w, w+8}; wave 0 runs a 3-accumulator variant that folds the FM tile.
  {
    int c0 = wv * 16 + mn, c1i = (wv + 8) * 16 + mn;
    const half_t* bp0 = cW1t + (size_t)c0 * DPAD;
    const half_t* bp1 = cW1t + (size_t)c1i * DPAD;
    float bias0 = cb1[c0], bias1 = cb1[c1i];
    v8f a0, a1;
    #pragma unroll
    for (int i = 0; i < 8; ++i) { a0[i] = bias0; a1[i] = bias1; }
    if (wv == 0) {                       // scalar branch, hoisted out of loop
      const half_t* bpF = fmWt + (size_t)mn * DPAD;
      v8f aF;
      #pragma unroll
      for (int i = 0; i < 8; ++i) aF[i] = 0.0f;
      for (int kb = 0; kb < 15; ++kb) {
        v16h a = load_a_frag(arow, DPAD, mn, grp, kb * 32);
        a0 = wmma16(a, load_b_row(bp0, grp, kb * 32), a0);
        a1 = wmma16(a, load_b_row(bp1, grp, kb * 32), a1);
        aF = wmma16(a, load_b_row(bpF, grp, kb * 32), aF);
      }
      #pragma unroll
      for (int i = 0; i < 8; ++i) lds_xe[i + grp * 8][mn] = aF[i];
    } else {
      for (int kb = 0; kb < 15; ++kb) {
        v16h a = load_a_frag(arow, DPAD, mn, grp, kb * 32);
        a0 = wmma16(a, load_b_row(bp0, grp, kb * 32), a0);
        a1 = wmma16(a, load_b_row(bp1, grp, kb * 32), a1);
      }
    }
    #pragma unroll
    for (int i = 0; i < 8; ++i) {
      int m = i + grp * 8;
      lds_c1[m][c0]  = (half_t)fmaxf(a0[i], 0.0f);
      lds_c1[m][c1i] = (half_t)fmaxf(a1[i], 0.0f);
    }
  }
  __syncthreads();

  // fm = 0.5*((sum xe)^2 - sum xe^2) per row (only first 10 cols are real)
  if (threadIdx.x < 16) {
    float s = 0.0f, sq = 0.0f;
    #pragma unroll
    for (int c = 0; c < 10; ++c) {
      float v = lds_xe[threadIdx.x][c];
      s += v; sq += v * v;
    }
    lds_fm[threadIdx.x] = 0.5f * (s * s - sq);
  }

  // GEMM2: c2[16,128] = relu(c1 @ cW2 + cb2), K=256; one tile per wave
  {
    int col = wv * 16 + mn;
    const half_t* bp = cW2t + (size_t)col * 256;
    float bias = cb2[col];
    v8f acc;
    #pragma unroll
    for (int i = 0; i < 8; ++i) acc[i] = bias;
    for (int kb = 0; kb < 8; ++kb) {
      v16h a = load_a_frag(&lds_c1[0][0], H1, mn, grp, kb * 32);
      acc = wmma16(a, load_b_row(bp, grp, kb * 32), acc);
    }
    #pragma unroll
    for (int i = 0; i < 8; ++i)
      lds_c2[i + grp * 8][col] = (half_t)fmaxf(acc[i], 0.0f);
  }
  __syncthreads();

  // GEMM3 + fusion: h = x + (c2 @ coW + cob) + fm, K=128.  coWt padded to
  // N=512 rows (zeros), so each wave register-blocks exactly 4 tiles.
  {
    v8f acc[4];
    const half_t* bp[4];
    float fmv[8];
    #pragma unroll
    for (int j = 0; j < 4; ++j) {
      int col = (wv * 4 + j) * 16 + mn;
      float bias = (col < DREAL) ? cob[col] : 0.0f;
      #pragma unroll
      for (int i = 0; i < 8; ++i) acc[j][i] = bias;
      bp[j] = coWt + (size_t)col * H2;
    }
    for (int kb = 0; kb < 4; ++kb) {
      v16h a = load_a_frag(&lds_c2[0][0], H2, mn, grp, kb * 32);
      #pragma unroll
      for (int j = 0; j < 4; ++j)
        acc[j] = wmma16(a, load_b_row(bp[j], grp, kb * 32), acc[j]);
    }
    #pragma unroll
    for (int i = 0; i < 8; ++i) fmv[i] = lds_fm[i + grp * 8];
    #pragma unroll
    for (int j = 0; j < 4; ++j) {
      int nt = wv * 4 + j;
      if (nt < 29) {                       // scalar store guard
        int col = nt * 16 + mn;
        #pragma unroll
        for (int i = 0; i < 8; ++i) {
          int m = i + grp * 8;
          float hv = acc[j][i] + (float)arow[(size_t)m * DPAD + col] + fmv[i];
          hh[(size_t)(b0 + m) * DPAD + col] = (half_t)hv;
        }
      }
    }
  }
  // zero hh K-pad cols 464..479 (256 threads cover 16x16)
  {
    int r = threadIdx.x >> 4, cc = threadIdx.x & 15;
    hh[(size_t)(b0 + r) * DPAD + DREAL + cc] = (half_t)0.0f;
  }
}

// ---------------------------------------------------------------------------
// Stage 3: experts + gates + mix + task heads.  Block = 16 rows, 8 waves.
// Expert GEMM1 (dominant: 960 WMMA/block) register-blocks 8 N-tiles (+gate
// tile for waves 0-2) on a single A-fragment load per K-step.
// ---------------------------------------------------------------------------
__global__ __launch_bounds__(256) void moe_head(
    const half_t* __restrict__ hh,
    const half_t* __restrict__ expW1t, const half_t* __restrict__ expW2t,
    const half_t* __restrict__ gateWt,
    const float* __restrict__ expb1, const float* __restrict__ expb2,
    const float* __restrict__ gateB, const float* __restrict__ taskW,
    const float* __restrict__ taskB, const float* __restrict__ logv,
    float* __restrict__ out) {
  __shared__ __align__(16) half_t lds_e1[16][NEXP * H1];  // relu expert layer1
  __shared__ float lds_eo[16][NEXP * H2];                 // expert outputs
  __shared__ float lds_gl[NTASK][16][16];                 // gate logits
  __shared__ float lds_g[16][NTASK][NEXP];                // softmaxed gates
  __shared__ float lds_s[16][NEXP][NTASK];                // eo . taskW

  const int b0   = blockIdx.x * 16;
  const int wv   = __builtin_amdgcn_readfirstlane(threadIdx.x) >> 5;
  const int lane = threadIdx.x & 31;
  const int grp  = lane >> 4;
  const int mn   = lane & 15;
  const half_t* arow = hh + (size_t)b0 * DPAD;

  // Expert GEMM1: e1[b,e,:] = relu(h @ expW1[e] + b1[e]).  Wave w owns tiles
  // 8w..8w+7 -> expert e = w>>1 fixed, 8 consecutive N-tiles.  Waves 0-2 run
  // a 9-accumulator loop variant that folds gate logits for task=w.
  {
    v8f acc[8];
    const half_t* bp[8];
    int ecol[8];
    #pragma unroll
    for (int j = 0; j < 8; ++j) {
      int tile = wv * 8 + j;
      int e = tile >> 4, nt = tile & 15;
      int c = e * H1 + nt * 16 + mn;          // row into [E*H1][DPAD]
      ecol[j] = c;
      float bias = expb1[c];
      #pragma unroll
      for (int i = 0; i < 8; ++i) acc[j][i] = bias;
      bp[j] = expW1t + (size_t)c * DPAD;
    }
    if (wv < NTASK) {                         // scalar branch, loop duplicated
      const half_t* gp = gateWt + (size_t)(wv * 16 + mn) * DPAD;
      float gbias = (mn < NEXP) ? gateB[wv * NEXP + mn] : 0.0f;
      v8f accG;
      #pragma unroll
      for (int i = 0; i < 8; ++i) accG[i] = gbias;
      for (int kb = 0; kb < 15; ++kb) {
        v16h a = load_a_frag(arow, DPAD, mn, grp, kb * 32);
        #pragma unroll
        for (int j = 0; j < 8; ++j)
          acc[j] = wmma16(a, load_b_row(bp[j], grp, kb * 32), acc[j]);
        accG = wmma16(a, load_b_row(gp, grp, kb * 32), accG);
      }
      #pragma unroll
      for (int i = 0; i < 8; ++i) lds_gl[wv][i + grp * 8][mn] = accG[i];
    } else {
      for (int kb = 0; kb < 15; ++kb) {
        v16h a = load_a_frag(arow, DPAD, mn, grp, kb * 32);
        #pragma unroll
        for (int j = 0; j < 8; ++j)
          acc[j] = wmma16(a, load_b_row(bp[j], grp, kb * 32), acc[j]);
      }
    }
    #pragma unroll
    for (int j = 0; j < 8; ++j) {
      #pragma unroll
      for (int i = 0; i < 8; ++i)
        lds_e1[i + grp * 8][ecol[j]] = (half_t)fmaxf(acc[j][i], 0.0f);
    }
  }
  __syncthreads();

  // softmax over experts per (row, task)
  if (threadIdx.x < 16 * NTASK) {
    int m = threadIdx.x / NTASK, task = threadIdx.x % NTASK;
    float l[NEXP], mx = -1e30f;
    #pragma unroll
    for (int e = 0; e < NEXP; ++e) { l[e] = lds_gl[task][m][e]; mx = fmaxf(mx, l[e]); }
    float den = 0.0f;
    #pragma unroll
    for (int e = 0; e < NEXP; ++e) { l[e] = __expf(l[e] - mx); den += l[e]; }
    float inv = 1.0f / den;
    #pragma unroll
    for (int e = 0; e < NEXP; ++e) lds_g[m][task][e] = l[e] * inv;
  }

  // Expert GEMM2: eo[b,e,:] = e1[b,e,:] @ expW2[e] + b2[e], K=256.
  // Wave w owns expert e = w>>1 (A reused) and 4 N-tiles.
  {
    int e = wv >> 1;
    const half_t* Ap = &lds_e1[0][0] + e * H1;   // row stride NEXP*H1
    v8f acc[4];
    const half_t* bp[4];
    int col[4];
    #pragma unroll
    for (int j = 0; j < 4; ++j) {
      int nt = (wv & 1) * 4 + j;
      col[j] = nt * 16 + mn;
      float bias = expb2[e * H2 + col[j]];
      #pragma unroll
      for (int i = 0; i < 8; ++i) acc[j][i] = bias;
      bp[j] = expW2t + (size_t)(e * H2 + col[j]) * 256;
    }
    for (int kb = 0; kb < 8; ++kb) {
      v16h a = load_a_frag(Ap, NEXP * H1, mn, grp, kb * 32);
      #pragma unroll
      for (int j = 0; j < 4; ++j)
        acc[j] = wmma16(a, load_b_row(bp[j], grp, kb * 32), acc[j]);
    }
    #pragma unroll
    for (int j = 0; j < 4; ++j) {
      #pragma unroll
      for (int i = 0; i < 8; ++i)
        lds_eo[i + grp * 8][e * H2 + col[j]] = acc[j][i];
    }
  }
  __syncthreads();

  // s[m][e][t] = eo[m,e,:] . taskW[t]  (192 dot products of length 128)
  if (threadIdx.x < 16 * NEXP * NTASK) {
    int m = threadIdx.x / (NEXP * NTASK);
    int r = threadIdx.x % (NEXP * NTASK);
    int e = r / NTASK, task = r % NTASK;
    float acc = 0.0f;
    #pragma unroll 4
    for (int o = 0; o < H2; ++o)
      acc = fmaf(lds_eo[m][e * H2 + o], taskW[task * H2 + o], acc);
    lds_s[m][e][task] = acc;
  }
  __syncthreads();

  // out[b][t] = sum_e g[b][t][e] * s[b][e][t] + task_b[t]
  if (threadIdx.x < 16 * NTASK) {
    int m = threadIdx.x / NTASK, task = threadIdx.x % NTASK;
    float o = taskB[task];
    #pragma unroll
    for (int e = 0; e < NEXP; ++e) o = fmaf(lds_g[m][task][e], lds_s[m][e][task], o);
    out[(size_t)(b0 + m) * NTASK + task] = o;
  }
  if (blockIdx.x == 0 && threadIdx.x < NTASK)
    out[(size_t)BATCH * NTASK + threadIdx.x] = logv[threadIdx.x];
}

// ---------------------------------------------------------------------------
extern "C" void kernel_launch(void* const* d_in, const int* in_sizes, int n_in,
                              void* d_out, int out_size, void* d_ws, size_t ws_size,
                              hipStream_t stream) {
  (void)in_sizes; (void)n_in; (void)out_size; (void)ws_size;
  const int*   disc  = (const int*)  d_in[0];
  const float* cont  = (const float*)d_in[1];
  const float* e0 = (const float*)d_in[2],  *e1 = (const float*)d_in[3];
  const float* e2 = (const float*)d_in[4],  *e3 = (const float*)d_in[5];
  const float* e4 = (const float*)d_in[6],  *e5 = (const float*)d_in[7];
  const float* e6 = (const float*)d_in[8],  *e7 = (const float*)d_in[9];
  const float* contW = (const float*)d_in[10];
  const float* contB = (const float*)d_in[11];
  const float* fmW   = (const float*)d_in[12];
  const float* cW1   = (const float*)d_in[13];
  const float* cb1   = (const float*)d_in[14];
  const float* cW2   = (const float*)d_in[15];
  const float* cb2   = (const float*)d_in[16];
  const float* coW   = (const float*)d_in[17];
  const float* cob   = (const float*)d_in[18];
  const float* expW1 = (const float*)d_in[19];
  const float* expb1 = (const float*)d_in[20];
  const float* expW2 = (const float*)d_in[21];
  const float* expb2 = (const float*)d_in[22];
  const float* gateW = (const float*)d_in[23];
  const float* gateB = (const float*)d_in[24];
  const float* taskW = (const float*)d_in[25];
  const float* taskB = (const float*)d_in[26];
  const float* logv  = (const float*)d_in[27];

  // Workspace carve-out (≈33 MB total), 256B aligned segments
  char* ws = (char*)d_ws;
  size_t off = 0;
  auto carve = [&](size_t bytes) -> void* {
    void* p = ws + off;
    off = (off + bytes + 255) & ~(size_t)255;
    return p;
  };
  half_t* xh     = (half_t*)carve((size_t)BATCH * DPAD * 2);
  half_t* hh     = (half_t*)carve((size_t)BATCH * DPAD * 2);
  half_t* cW1t   = (half_t*)carve((size_t)H1 * DPAD * 2);
  half_t* cW2t   = (half_t*)carve((size_t)H2 * 256 * 2);
  half_t* coWt   = (half_t*)carve((size_t)COPAD * H2 * 2);   // N padded 464->512
  half_t* fmWt   = (half_t*)carve((size_t)16 * DPAD * 2);
  half_t* expW1t = (half_t*)carve((size_t)NEXP * H1 * DPAD * 2);
  half_t* expW2t = (half_t*)carve((size_t)NEXP * H2 * 256 * 2);
  half_t* gateWt = (half_t*)carve((size_t)NTASK * 16 * DPAD * 2);

  auto conv = [&](const float* src, half_t* dst, int batch, int Nd, int Ns,
                  int Kpad, int Ks) {
    int total = batch * Nd * Kpad;
    convert_weights<<<(total + 255) / 256, 256, 0, stream>>>(src, dst, Nd, Ns,
                                                             Kpad, Ks, total);
  };
  conv(cW1,   cW1t,   1,     H1,    H1,    DPAD, DREAL); // [464,256]->[256][480]
  conv(cW2,   cW2t,   1,     H2,    H2,    256,  H1);    // [256,128]->[128][256]
  conv(coW,   coWt,   1,     COPAD, DREAL, H2,   H2);    // [128,464]->[512][128]
  conv(fmW,   fmWt,   1,     16,    10,    DPAD, DREAL); // [464,10] ->[16][480]
  conv(expW1, expW1t, NEXP,  H1,    H1,    DPAD, DREAL); // [4,464,256]
  conv(expW2, expW2t, NEXP,  H2,    H2,    256,  H1);    // [4,256,128]
  conv(gateW, gateWt, NTASK, 16,    NEXP,  DPAD, DREAL); // [3,464,4]->[3][16][480]

  build_x<<<BATCH, 128, 0, stream>>>(disc, cont, e0, e1, e2, e3, e4, e5, e6, e7,
                                     contW, contB, xh);
  cross_fm<<<BATCH / 16, 256, 0, stream>>>(xh, cb1, cb2, cob, cW1t, cW2t, coWt,
                                           fmWt, hh);
  moe_head<<<BATCH / 16, 256, 0, stream>>>(hh, expW1t, expW2t, gateWt, expb1,
                                           expb2, gateB, taskW, taskB, logv,
                                           (float*)d_out);
}
